// RNNAE_24215025615651
// MI455X (gfx1250) — matrix operand order repaired and checked by reading, not compiled, not run on hardware
//
#include <hip/hip_runtime.h>
#include <stdint.h>

// Problem dimensions (match reference)
#define Bn    64
#define Tn    512
#define Dn    128
#define Hn    512
#define RECL  384          // T - T/4
#define PREDL 128
#define G4H   (4 * Hn)
#define BH    (Bn * Hn)
#define BD    (Bn * Dn)
#define KC    128          // K-panel staged through LDS; divides K1 (128/512) and Hn

typedef __attribute__((ext_vector_type(16))) __bf16 v16bf;
typedef __attribute__((ext_vector_type(8)))  float  v8f;

// ---------------- helpers ----------------

__device__ __forceinline__ unsigned short f2bf(float f) {
  union { float f; uint32_t u; } v; v.f = f;
  uint32_t u = v.u;
  uint32_t r = (u + 0x7FFFu + ((u >> 16) & 1u)) >> 16;            // RNE
  if ((u & 0x7F800000u) == 0x7F800000u)                            // inf/nan
    r = (u >> 16) | ((u & 0xFFFFu) ? 0x40u : 0u);
  return (unsigned short)r;
}

// Branchless activations: v_exp + v_rcp, no EXEC-divergent libm paths.
__device__ __forceinline__ float sigm(float x) {
  return __builtin_amdgcn_rcpf(1.0f + __expf(-x));
}
__device__ __forceinline__ float fast_tanh(float x) {
  float e = __expf(-2.0f * fabsf(x));
  float t = (1.0f - e) * __builtin_amdgcn_rcpf(1.0f + e);
  return copysignf(t, x);
}

union ABu { uint4 q[2]; v16bf v; };

// A fragment: 16x32 bf16 tile from a per-chunk row base pointer (base = row start + k0).
// Layout used here: lane l -> row l&15; kb = (l>>4)*16; K chunk kb..kb+15 (contiguous, matches B).
// kk is a compile-time-unrolled multiple of 32, so it folds into the load's
// immediate offset — no per-load address VALU.
__device__ __forceinline__ v16bf load_a(const unsigned short* __restrict__ base,
                                        int kk, int lane) {
  int kb = kk + ((lane >> 4) << 4);
  ABu u;
  u.q[0] = *(const uint4*)(base + kb);
  u.q[1] = *(const uint4*)(base + kb + 8);
  return u.v;
}

// ---------------- fused LSTM cell step ----------------
// gates = [x_t | h_in] @ Wcat^T + bsum ; then i,f,g,o activations; update c; h_out = o*tanh(c)
// grid: Hn/16 = 32 blocks (one 16-wide hidden slice each), block: 128 threads = 4 waves.
// Each wave owns a 16-row batch tile; computes 4 gate tiles (i,f,g,o) for the slice.
// B panel (4 gates x 16 cols x KC) is staged through LDS once per chunk and shared
// by all 4 waves (4x less L2 traffic; inner loop feeds WMMA from ds_load).
// K1 is a multiple of KC, so each K-chunk's A source is uniformly x or h.
__global__ __launch_bounds__(128)
void lstm_step(const unsigned short* __restrict__ xin, int K1,
               const unsigned short* __restrict__ h_in,
               unsigned short* __restrict__ h_out,
               float* __restrict__ c,
               const unsigned short* __restrict__ Wcat,   // (4H) x (K1+Hn) bf16
               const float* __restrict__ bsum)            // 4H (b_ih + b_hh)
{
  __shared__ __align__(32) unsigned short ldsB[64 * KC];   // 16 KB
  const int Kt      = K1 + Hn;
  const int tid     = threadIdx.x;
  const int lane    = tid & 31;
  const int wave    = tid >> 5;
  const int j0      = blockIdx.x * 16;
  const int rowBase = wave * 16;
  const int row     = rowBase + (lane & 15);
  const int col     = lane & 15;

  v8f acc[4];
#pragma unroll
  for (int g = 0; g < 4; ++g) {
    float bv = bsum[g * Hn + j0 + col];
#pragma unroll
    for (int i = 0; i < 8; ++i) acc[g][i] = bv;
  }

  for (int k0 = 0; k0 < Kt; k0 += KC) {
    // cooperative stage: 64 weight rows x KC elems = 1024 uint4, 8 per thread
#pragma unroll
    for (int it = 0; it < 8; ++it) {
      int linear = tid + 128 * it;        // uint4 index; 16 uint4 per row
      int ridx   = linear >> 4;           // 0..63 : g*16 + col
      int q      = linear & 15;
      int g      = ridx >> 4;
      int cc     = ridx & 15;
      const unsigned short* src = Wcat + (size_t)(g * Hn + j0 + cc) * Kt + k0 + q * 8;
      *(uint4*)&ldsB[ridx * KC + q * 8] = *(const uint4*)src;
    }
    __syncthreads();

    // uniform per-chunk A segment select (k0, K1 uniform -> scalar branch)
    const unsigned short* aseg = (k0 < K1)
        ? (xin  + (size_t)row * K1 + k0)
        : (h_in + (size_t)row * Hn + (k0 - K1));

#pragma unroll
    for (int kk = 0; kk < KC; kk += 32) {
      v16bf a = load_a(aseg, kk, lane);
      int kb = kk + ((lane >> 4) << 4);   // B frag: lane col, 16 contiguous K from LDS
#pragma unroll
      for (int g = 0; g < 4; ++g) {
        v16bf b = *(const v16bf*)&ldsB[(g * 16 + col) * KC + kb];
        acc[g] = __builtin_amdgcn_wmma_f32_16x16x32_bf16(
            false, a, false, b, (short)0, acc[g], false, false);
      }
    }
    __syncthreads();
  }

  // D layout: vgpr i, lane l -> (m = rowBase + (l>>4)*8 + i, n = j0 + (l&15))
  const int n    = j0 + col;
  const int mOff = rowBase + ((lane >> 4) << 3);
#pragma unroll
  for (int i = 0; i < 8; ++i) {
    int m = mOff + i;
    float ig = sigm(acc[0][i]);
    float fg = sigm(acc[1][i]);
    float gg = fast_tanh(acc[2][i]);
    float og = sigm(acc[3][i]);
    size_t idx = (size_t)m * Hn + n;
    float cv = fg * c[idx] + ig * gg;     // this block exclusively owns column n of c
    c[idx] = cv;
    h_out[idx] = f2bf(og * fast_tanh(cv));
  }
}

// ---------------- decoder output projection ----------------
// out_t = h @ W^T + b  (M=64, N=128, K=512); writes f32 into out[:, t, :] and bf16 into ibuf.
// 1 block, 128 threads: each wave owns a 16-row batch tile x all 8 N-tiles.
__global__ __launch_bounds__(128)
void linear_out(const unsigned short* __restrict__ h,    // Bn x Hn bf16
                const unsigned short* __restrict__ W,    // Dn x Hn bf16
                const float* __restrict__ b,             // Dn
                float* __restrict__ out, int t,
                unsigned short* __restrict__ ibuf)       // Bn x Dn bf16
{
  __shared__ __align__(32) unsigned short ldsW[Dn * KC];  // 32 KB
  const int tid     = threadIdx.x;
  const int lane    = tid & 31;
  const int wave    = tid >> 5;
  const int rowBase = wave * 16;
  const int row     = rowBase + (lane & 15);
  const int col     = lane & 15;

  v8f acc[8];
#pragma unroll
  for (int nt = 0; nt < 8; ++nt) {
    float bv = b[nt * 16 + col];
#pragma unroll
    for (int i = 0; i < 8; ++i) acc[nt][i] = bv;
  }

  for (int k0 = 0; k0 < Hn; k0 += KC) {
    // stage 128 rows x KC elems = 2048 uint4, 16 per thread
#pragma unroll
    for (int it = 0; it < 16; ++it) {
      int linear = tid + 128 * it;
      int ridx   = linear >> 4;
      int q      = linear & 15;
      *(uint4*)&ldsW[ridx * KC + q * 8] =
          *(const uint4*)(W + (size_t)ridx * Hn + k0 + q * 8);
    }
    __syncthreads();

    const unsigned short* aseg = h + (size_t)row * Hn + k0;

#pragma unroll
    for (int kk = 0; kk < KC; kk += 32) {
      v16bf a = load_a(aseg, kk, lane);
      int kb = kk + ((lane >> 4) << 4);
#pragma unroll
      for (int nt = 0; nt < 8; ++nt) {
        v16bf bb = *(const v16bf*)&ldsW[(nt * 16 + col) * KC + kb];
        acc[nt] = __builtin_amdgcn_wmma_f32_16x16x32_bf16(
            false, a, false, bb, (short)0, acc[nt], false, false);
      }
    }
    __syncthreads();
  }

  const int mOff = rowBase + ((lane >> 4) << 3);
#pragma unroll
  for (int nt = 0; nt < 8; ++nt) {
    int n = nt * 16 + col;
#pragma unroll
    for (int i = 0; i < 8; ++i) {
      int m = mOff + i;
      float v = acc[nt][i];
      out[((size_t)m * Tn + t) * Dn + n] = v;
      ibuf[(size_t)m * Dn + n] = f2bf(v);
    }
  }
}

// ---------------- packing / init kernels ----------------

// Wcat[r][k] = bf16( k<K1 ? Wih[r][k] : Whh[r][k-K1] );  bsum[r] = bih[r]+bhh[r]
__global__ void pack_cell(const float* __restrict__ Wih, const float* __restrict__ Whh,
                          const float* __restrict__ bih, const float* __restrict__ bhh,
                          unsigned short* __restrict__ Wcat, float* __restrict__ bsum,
                          int K1) {
  int Kt = K1 + Hn;
  long total = (long)G4H * Kt;
  long stride = (long)gridDim.x * blockDim.x;
  for (long idx = (long)blockIdx.x * blockDim.x + threadIdx.x; idx < total; idx += stride) {
    int r = (int)(idx / Kt), k = (int)(idx % Kt);
    float v = (k < K1) ? Wih[(long)r * K1 + k] : Whh[(long)r * Hn + (k - K1)];
    Wcat[idx] = f2bf(v);
  }
  for (long r = (long)blockIdx.x * blockDim.x + threadIdx.x; r < G4H; r += stride)
    bsum[r] = bih[r] + bhh[r];
}

__global__ void conv_bf16(const float* __restrict__ src, unsigned short* __restrict__ dst, long n) {
  long stride = (long)gridDim.x * blockDim.x;
  for (long i = (long)blockIdx.x * blockDim.x + threadIdx.x; i < n; i += stride)
    dst[i] = f2bf(src[i]);
}

// xr[t][b][d] = bf16( x[b][RECL-1-t][d] )   (truncate + time-flip + time-major)
__global__ void pack_xrev(const float* __restrict__ x, unsigned short* __restrict__ xr) {
  long total = (long)RECL * Bn * Dn;
  long stride = (long)gridDim.x * blockDim.x;
  for (long idx = (long)blockIdx.x * blockDim.x + threadIdx.x; idx < total; idx += stride) {
    int d = (int)(idx % Dn);
    int b = (int)((idx / Dn) % Bn);
    int t = (int)(idx / ((long)Dn * Bn));
    xr[idx] = f2bf(x[((long)b * Tn + (RECL - 1 - t)) * Dn + d]);
  }
}

__global__ void zero_u32(uint32_t* __restrict__ p, long n) {
  long stride = (long)gridDim.x * blockDim.x;
  for (long i = (long)blockIdx.x * blockDim.x + threadIdx.x; i < n; i += stride)
    p[i] = 0u;
}

// ---------------- host orchestration ----------------

extern "C" void kernel_launch(void* const* d_in, const int* in_sizes, int n_in,
                              void* d_out, int out_size, void* d_ws, size_t ws_size,
                              hipStream_t stream) {
  (void)in_sizes; (void)n_in; (void)out_size; (void)ws_size;
  const float* x      = (const float*)d_in[0];
  const float* eWih0  = (const float*)d_in[1];
  const float* eWhh0  = (const float*)d_in[2];
  const float* ebih0  = (const float*)d_in[3];
  const float* ebhh0  = (const float*)d_in[4];
  const float* eWih1  = (const float*)d_in[5];
  const float* eWhh1  = (const float*)d_in[6];
  const float* ebih1  = (const float*)d_in[7];
  const float* ebhh1  = (const float*)d_in[8];
  const float* rWih   = (const float*)d_in[9];
  const float* rWhh   = (const float*)d_in[10];
  const float* rbih   = (const float*)d_in[11];
  const float* rbhh   = (const float*)d_in[12];
  const float* pWih   = (const float*)d_in[13];
  const float* pWhh   = (const float*)d_in[14];
  const float* pbih   = (const float*)d_in[15];
  const float* pbhh   = (const float*)d_in[16];
  const float* lrW    = (const float*)d_in[17];
  const float* lrb    = (const float*)d_in[18];
  const float* lpW    = (const float*)d_in[19];
  const float* lpb    = (const float*)d_in[20];
  float* out = (float*)d_out;

  char* ws = (char*)d_ws;
  size_t off = 0;
  auto alloc = [&](size_t bytes) -> void* {
    void* p = ws + off;
    off = (off + bytes + 255) & ~(size_t)255;
    return p;
  };

  unsigned short* Wcat0 = (unsigned short*)alloc((size_t)G4H * (Dn + Hn) * 2);
  float*          bs0   = (float*)alloc(G4H * 4);
  unsigned short* Wcat1 = (unsigned short*)alloc((size_t)G4H * (Hn + Hn) * 2);
  float*          bs1   = (float*)alloc(G4H * 4);
  unsigned short* WcatR = (unsigned short*)alloc((size_t)G4H * (Dn + Hn) * 2);
  float*          bsR   = (float*)alloc(G4H * 4);
  unsigned short* WcatP = (unsigned short*)alloc((size_t)G4H * (Dn + Hn) * 2);
  float*          bsP   = (float*)alloc(G4H * 4);
  unsigned short* lrW16 = (unsigned short*)alloc((size_t)Dn * Hn * 2);
  unsigned short* lpW16 = (unsigned short*)alloc((size_t)Dn * Hn * 2);
  unsigned short* xrev  = (unsigned short*)alloc((size_t)RECL * BD * 2);
  unsigned short* ys    = (unsigned short*)alloc((size_t)RECL * BH * 2);  // layer0 h history (bf16)
  unsigned short* hz    = (unsigned short*)alloc((size_t)BH * 2);         // zero h
  unsigned short* hA    = (unsigned short*)alloc((size_t)BH * 2);
  unsigned short* hB    = (unsigned short*)alloc((size_t)BH * 2);
  unsigned short* hdA   = (unsigned short*)alloc((size_t)BH * 2);
  unsigned short* hdB   = (unsigned short*)alloc((size_t)BH * 2);
  float*          c0    = (float*)alloc((size_t)BH * 4);
  float*          c1    = (float*)alloc((size_t)BH * 4);
  float*          cd    = (float*)alloc((size_t)BH * 4);
  unsigned short* ibuf  = (unsigned short*)alloc((size_t)BD * 2);

  // ---- pack weights / inputs, zero initial states ----
  pack_cell<<<512, 256, 0, stream>>>(eWih0, eWhh0, ebih0, ebhh0, Wcat0, bs0, Dn);
  pack_cell<<<512, 256, 0, stream>>>(eWih1, eWhh1, ebih1, ebhh1, Wcat1, bs1, Hn);
  pack_cell<<<512, 256, 0, stream>>>(rWih, rWhh, rbih, rbhh, WcatR, bsR, Dn);
  pack_cell<<<512, 256, 0, stream>>>(pWih, pWhh, pbih, pbhh, WcatP, bsP, Dn);
  conv_bf16<<<256, 256, 0, stream>>>(lrW, lrW16, (long)Dn * Hn);
  conv_bf16<<<256, 256, 0, stream>>>(lpW, lpW16, (long)Dn * Hn);
  pack_xrev<<<512, 256, 0, stream>>>(x, xrev);
  zero_u32<<<256, 256, 0, stream>>>((uint32_t*)hz, BH / 2);
  zero_u32<<<256, 256, 0, stream>>>((uint32_t*)c0, BH);
  zero_u32<<<256, 256, 0, stream>>>((uint32_t*)c1, BH);

  const int NBLK = Hn / 16;   // 32 tile blocks per step

  // ---- encoder layer 0: h history lands directly in ys ----
  for (int t = 0; t < RECL; ++t) {
    const unsigned short* hin = (t == 0) ? hz : (ys + (size_t)(t - 1) * BH);
    lstm_step<<<NBLK, 128, 0, stream>>>(xrev + (size_t)t * BD, Dn,
                                        hin, ys + (size_t)t * BH, c0, Wcat0, bs0);
  }

  // ---- encoder layer 1: ping-pong h, keep only final (h2, c2) ----
  const unsigned short* h2 = hz;
  for (int t = 0; t < RECL; ++t) {
    unsigned short* hout = (t & 1) ? hB : hA;
    lstm_step<<<NBLK, 128, 0, stream>>>(ys + (size_t)t * BH, Hn,
                                        h2, hout, c1, Wcat1, bs1);
    h2 = hout;
  }

  // ---- decoders: emit linear(h) first, then (cell; linear) x (n-1) ----
  auto run_decoder = [&](const unsigned short* Wcat, const float* bs,
                         const unsigned short* Wl, const float* bl,
                         int tbase, int nsteps) {
    hipMemcpyAsync(hdA, (const void*)h2, (size_t)BH * 2, hipMemcpyDeviceToDevice, stream);
    hipMemcpyAsync(cd, (const void*)c1, (size_t)BH * 4, hipMemcpyDeviceToDevice, stream);
    const unsigned short* hcur = hdA;
    linear_out<<<1, 128, 0, stream>>>(hcur, Wl, bl, out, tbase, ibuf);
    for (int s = 1; s < nsteps; ++s) {
      unsigned short* hout = (s & 1) ? hdB : hdA;
      lstm_step<<<NBLK, 128, 0, stream>>>(ibuf, Dn, hcur, hout, cd, Wcat, bs);
      hcur = hout;
      linear_out<<<1, 128, 0, stream>>>(hcur, Wl, bl, out, tbase + s, ibuf);
    }
  };
  run_decoder(WcatR, bsR, lrW16, lrb, 0, RECL);
  run_decoder(WcatP, bsP, lpW16, lpb, RECL, PREDL);
}
